// HybridTransformerLayer_74491912782025
// MI455X (gfx1250) — matrix operand order repaired and checked by reading, not compiled, not run on hardware
//
#include <hip/hip_runtime.h>
#include <stdint.h>

// ---------------------------------------------------------------------------
// Types
// ---------------------------------------------------------------------------
typedef _Float16 half_t;
typedef __attribute__((ext_vector_type(16))) _Float16 v16h;
typedef __attribute__((ext_vector_type(8)))  _Float16 v8h;
typedef __attribute__((ext_vector_type(8)))  float    v8f;
typedef __attribute__((ext_vector_type(4)))  unsigned int u32x4;
typedef __attribute__((ext_vector_type(8)))  int i32x8;
typedef __attribute__((ext_vector_type(4)))  int i32x4;

#define NNODES 4096
#define HID    512
#define HEADS  8
#define DH     64
#define NC     64
#define NEDGES 65536
#define MASKW  128            // 4096 bits / 32
#define ATT_SCALE 0.125f      // 1/sqrt(64)

// ---- Tensor Data Mover availability probes --------------------------------
#if defined(__has_builtin)
#if __has_builtin(__builtin_amdgcn_tensor_load_to_lds)
#define HAS_TDM 1
#endif
#if __has_builtin(__builtin_amdgcn_s_wait_tensorcnt)
#define HAS_WAIT_TENSOR 1
#endif
#endif
#ifndef HAS_TDM
#define HAS_TDM 0
#endif
#ifndef HAS_WAIT_TENSOR
#define HAS_WAIT_TENSOR 0
#endif

__device__ __forceinline__ v16h frag_cat(v8h lo, v8h hi) {
  return __builtin_shufflevector(lo, hi, 0,1,2,3,4,5,6,7,8,9,10,11,12,13,14,15);
}
__device__ __forceinline__ v8f wmma_f16(v16h a, v16h b, v8f c) {
  // D = A(16x32 f16) * B(32x16 f16) + C(16x16 f32)
  return __builtin_amdgcn_wmma_f32_16x16x32_f16(false, a, false, b, (short)0, c,
                                                false, false);
}

#if HAS_TDM
// 2D tensor tile -> LDS via TDM.  dims/strides in 2-byte elements.
// Writes tile rows of tile_d0 halves with 16-half (32B) padding after each
// 32-DWORD (128B) row => LDS row stride 160B (matches STRIDE=80 below).
__device__ __forceinline__ void tdm_load_2d(uint32_t ldsOff, const void* gptr,
                                            uint32_t tensor_d0,
                                            uint32_t tensor_d1,
                                            uint64_t stride0,
                                            uint32_t tile_d0,
                                            uint32_t tile_d1) {
  uint64_t ga = (uint64_t)(uintptr_t)gptr;
  u32x4 g0;
  g0[0] = 1u;                                   // count=1, user descriptor
  g0[1] = ldsOff;                               // lds_addr (bytes)
  g0[2] = (uint32_t)ga;                         // global_addr[31:0]
  g0[3] = (uint32_t)((ga >> 32) & 0x01FFFFFFu)  // global_addr[56:32]
          | (2u << 30);                         // type=2 ("image")
  i32x8 g1;
  // data_size=1 (2B) | pad_enable | pad_interval=4 (32 DW) | pad_amount=7 (8 DW)
  g1[0] = (int)((1u << 16) | (1u << 20) | (4u << 22) | (7u << 25));
  g1[1] = (int)((tensor_d0 & 0xFFFFu) << 16);               // dim0[15:0]
  g1[2] = (int)((tensor_d0 >> 16) | ((tensor_d1 & 0xFFFFu) << 16));
  g1[3] = (int)((tensor_d1 >> 16) | ((tile_d0 & 0xFFFFu) << 16));
  g1[4] = (int)(tile_d1 & 0xFFFFu);                         // tile_dim1 (tile_dim2=0)
  g1[5] = (int)(uint32_t)stride0;                           // dim0_stride[31:0]
  g1[6] = (int)(uint32_t)((stride0 >> 32) & 0xFFFFu);       // dim0_stride[47:32]
  g1[7] = 0;
  i32x4 z4 = {0, 0, 0, 0};
#if __clang_major__ >= 23
  i32x8 z8 = {0, 0, 0, 0, 0, 0, 0, 0};
  __builtin_amdgcn_tensor_load_to_lds(g0, g1, z4, z4, z8, 0);
#else
  __builtin_amdgcn_tensor_load_to_lds(g0, g1, z4, z4, 0);
#endif
}
__device__ __forceinline__ void tdm_wait() {
#if HAS_WAIT_TENSOR
  __builtin_amdgcn_s_wait_tensorcnt(0);
#else
  asm volatile("s_wait_tensorcnt 0x0" ::: "memory");
#endif
}
#endif  // HAS_TDM

// ---------------------------------------------------------------------------
// Bitset adjacency / 2-hop reachability (replaces 4096^3 f32 mask matmul)
// ---------------------------------------------------------------------------
__global__ void k_zero_u32(uint32_t* p, int n) {
  for (int i = blockIdx.x * blockDim.x + threadIdx.x; i < n;
       i += gridDim.x * blockDim.x)
    p[i] = 0u;
}

__global__ void k_edges(const int* __restrict__ ei, uint32_t* __restrict__ adj) {
  int e = blockIdx.x * blockDim.x + threadIdx.x;
  if (e >= NEDGES) return;
  int s = ei[e];
  int d = ei[NEDGES + e];
  atomicOr(&adj[(size_t)s * MASKW + (d >> 5)], 1u << (d & 31));
  atomicOr(&adj[(size_t)d * MASKW + (s >> 5)], 1u << (s & 31));
}

__global__ void k_diag(uint32_t* __restrict__ adj) {
  int i = blockIdx.x * blockDim.x + threadIdx.x;
  if (i >= NNODES) return;
  atomicOr(&adj[(size_t)i * MASKW + (i >> 5)], 1u << (i & 31));
}

__global__ __launch_bounds__(128) void k_reach(const uint32_t* __restrict__ adj,
                                               uint32_t* __restrict__ mask) {
  const int row = blockIdx.x;
  const int tid = threadIdx.x;          // 0..127, one word per thread
  uint32_t acc = 0u;
  const uint32_t* rp = adj + (size_t)row * MASKW;
  for (int j = 0; j < MASKW; ++j) {
    uint32_t u = rp[j];                  // broadcast load
    while (u) {
      int b = __ffs(u) - 1;
      u &= (u - 1);
      int k = (j << 5) + b;
      acc |= adj[(size_t)k * MASKW + tid];
    }
  }
  mask[(size_t)row * MASKW + tid] = acc;
}

// ---------------------------------------------------------------------------
// f32 -> f16 convert; f32 [K][N] -> f16 [N][K] transpose-convert
// ---------------------------------------------------------------------------
__global__ void k_cvt16(const float* __restrict__ in, half_t* __restrict__ out,
                        long n) {
  for (long i = blockIdx.x * blockDim.x + threadIdx.x; i < n;
       i += (long)gridDim.x * blockDim.x)
    out[i] = (half_t)in[i];
}

__global__ void k_transpose16(const float* __restrict__ W,
                              half_t* __restrict__ Wt, int K, int N) {
  long total = (long)K * N;
  for (long i = blockIdx.x * blockDim.x + threadIdx.x; i < total;
       i += (long)gridDim.x * blockDim.x) {
    int k = (int)(i / N);
    int n = (int)(i % N);
    Wt[(size_t)n * K + k] = (half_t)W[i];
  }
}

// ---------------------------------------------------------------------------
// WMMA GEMM:  C[M][N] = act(A[M][K](f16) * Bt[N][K](f16)^T + bias)
// Tile 128x64, 8 waves (256 threads), BK=64.  LDS staged by TDM when present.
// ---------------------------------------------------------------------------
template <bool RELU, bool OUT16>
__global__ __launch_bounds__(256) void k_gemm(
    const half_t* __restrict__ A, const half_t* __restrict__ Bt,
    const float* __restrict__ bias, float* __restrict__ C32,
    half_t* __restrict__ C16, int M, int N, int K, int lda, int ldb, int ldc) {
  constexpr int BM = 128, BN = 64, BK = 64, STRIDE = 80;  // 160B rows, 16B aligned
  __shared__ __align__(16) half_t Ash[BM][STRIDE];
  __shared__ __align__(16) half_t Bsh[BN][STRIDE];

  const int tid  = threadIdx.x;
  const int lane = tid & 31;
  const int wave = tid >> 5;
  const int wm   = (wave & 3) * 32;
  const int wn   = (wave >> 2) * 32;
  const int n16  = lane & 15;
  const int g    = lane >> 4;

  const int rowBase = blockIdx.y * BM;
  const int colBase = blockIdx.x * BN;

  v8f acc[2][2] = {};

#if HAS_TDM
  const uint32_t ldsA = (uint32_t)(uintptr_t)&Ash[0][0];
  const uint32_t ldsB = (uint32_t)(uintptr_t)&Bsh[0][0];
#else
  const int c8 = (tid & 7) * 8;          // 0..56 halves
  const int rA = tid >> 3;               // 0..31
#endif

  for (int k0 = 0; k0 < K; k0 += BK) {
#if HAS_TDM
    if (tid < 32) {  // wave-uniform: only wave 0 issues the two DMAs
      tdm_load_2d(ldsA, A + (size_t)rowBase * lda + k0,
                  (uint32_t)(K - k0), (uint32_t)(M - rowBase),
                  (uint64_t)lda, BK, BM);
      tdm_load_2d(ldsB, Bt + (size_t)colBase * ldb + k0,
                  (uint32_t)(K - k0), (uint32_t)(N - colBase),
                  (uint64_t)ldb, BK, BN);
      tdm_wait();
    }
    __syncthreads();
#else
#pragma unroll
    for (int p = 0; p < 4; ++p) {
      int r = rA + p * 32;
      int grow = rowBase + r;
      v8h v = {};
      if (grow < M) {
        const half_t* src = A + (size_t)grow * lda + k0 + c8;
        v = *(const v8h*)src;
        if (k0 + BK < K) __builtin_prefetch(src + BK, 0, 1);
      }
      *(v8h*)&Ash[r][c8] = v;
    }
#pragma unroll
    for (int p = 0; p < 2; ++p) {
      int r = rA + p * 32;
      const half_t* src = Bt + (size_t)(colBase + r) * ldb + k0 + c8;
      v8h v = *(const v8h*)src;
      if (k0 + BK < K) __builtin_prefetch(src + BK, 0, 1);
      *(v8h*)&Bsh[r][c8] = v;
    }
    __syncthreads();
#endif

#pragma unroll
    for (int s = 0; s < 2; ++s) {
      v16h af[2], bf[2];
#pragma unroll
      for (int i = 0; i < 2; ++i) {
        const half_t* p = &Ash[wm + i * 16 + n16][s * 32];
        af[i] = frag_cat(*(const v8h*)(p + 8 * g), *(const v8h*)(p + 16 + 8 * g));
      }
#pragma unroll
      for (int j = 0; j < 2; ++j) {
        const half_t* p = &Bsh[wn + j * 16 + n16][s * 32];
        bf[j] = frag_cat(*(const v8h*)(p + 8 * g), *(const v8h*)(p + 16 + 8 * g));
      }
#pragma unroll
      for (int i = 0; i < 2; ++i)
#pragma unroll
        for (int j = 0; j < 2; ++j)
          acc[i][j] = wmma_f16(af[i], bf[j], acc[i][j]);
    }
    __syncthreads();
  }

#pragma unroll
  for (int i = 0; i < 2; ++i) {
#pragma unroll
    for (int j = 0; j < 2; ++j) {
      int col = colBase + wn + j * 16 + n16;
      float bv = bias ? bias[col] : 0.0f;
#pragma unroll
      for (int r = 0; r < 8; ++r) {
        int row = rowBase + wm + i * 16 + r + 8 * g;
        if (row < M) {
          float v = acc[i][j][r] + bv;
          if (RELU) v = v > 0.0f ? v : 0.0f;
          if (OUT16) C16[(size_t)row * ldc + col] = (half_t)v;
          else       C32[(size_t)row * ldc + col] = v;
        }
      }
    }
  }
}

// ---------------------------------------------------------------------------
// Flash attention: 4 waves/block, each wave one 16-query tile, all sharing the
// cooperative V^T staging (amortizes the LDS transpose 4x).
// ---------------------------------------------------------------------------
__global__ __launch_bounds__(128) void k_flash(
    const half_t* __restrict__ Q, const half_t* __restrict__ Kp,
    const half_t* __restrict__ Vp, const uint32_t* __restrict__ maskBits,
    half_t* __restrict__ Out, int nKeys, int maskStride, float scale) {
  constexpr int LD = HID;
  __shared__ __align__(16) half_t Vt[64][32];       // [dh][key] shared by block
  __shared__ __align__(16) half_t Pt[4][16][32];    // per-wave P tiles

  const int tid  = threadIdx.x;
  const int lane = tid & 31;
  const int wave = tid >> 5;
  const int n16  = lane & 15;
  const int g    = lane >> 4;
  const int head = blockIdx.y;
  const int qBase = blockIdx.x * 64 + wave * 16;
  const int hOff  = head * DH;

  v16h qf[2];
#pragma unroll
  for (int kk = 0; kk < 2; ++kk) {
    const half_t* p = Q + (size_t)(qBase + n16) * LD + hOff + kk * 32;
    qf[kk] = frag_cat(*(const v8h*)(p + 16 * g), *(const v8h*)(p + 16 * g + 8));
  }

  float mrow[8], lrow[8];
  v8f o[4] = {};
#pragma unroll
  for (int r = 0; r < 8; ++r) { mrow[r] = -3.0e38f; lrow[r] = 0.0f; }

  for (int kb = 0; kb < nKeys; kb += 32) {
    // cooperative V^T staging: 128 threads, 4 threads per key row,
    // each covers a 16-wide dh chunk.
    {
      const int kcol = tid >> 2;               // 0..31
      const int dh0  = (tid & 3) * 16;         // 0,16,32,48
      const half_t* vp = Vp + (size_t)(kb + kcol) * LD + hOff + dh0;
      v8h a = *(const v8h*)vp;
      v8h b = *(const v8h*)(vp + 8);
#pragma unroll
      for (int e = 0; e < 8; ++e) Vt[dh0 + e][kcol] = a[e];
#pragma unroll
      for (int e = 0; e < 8; ++e) Vt[dh0 + 8 + e][kcol] = b[e];
    }

    // S = Q * K^T for 32 keys (two 16-col fragments)
    v8f s0 = {}, s1 = {};
#pragma unroll
    for (int kk = 0; kk < 2; ++kk) {
      const half_t* p0 = Kp + (size_t)(kb + n16) * LD + hOff + kk * 32 + 16 * g;
      const half_t* p1 = Kp + (size_t)(kb + 16 + n16) * LD + hOff + kk * 32 + 16 * g;
      v16h b0 = frag_cat(*(const v8h*)p0, *(const v8h*)(p0 + 8));
      v16h b1 = frag_cat(*(const v8h*)p1, *(const v8h*)(p1 + 8));
      s0 = wmma_f16(qf[kk], b0, s0);
      s1 = wmma_f16(qf[kk], b1, s1);
    }

    // online softmax (row m = r + 8g lives in the 16 lanes of group g)
    float corr[8];
#pragma unroll
    for (int r = 0; r < 8; ++r) {
      float a0 = s0[r] * scale, a1 = s1[r] * scale;
      float keep0 = 1.0f, keep1 = 1.0f;
      if (maskBits) {
        uint32_t w =
            maskBits[(size_t)(qBase + r + 8 * g) * maskStride + (kb >> 5)];
        if (!((w >> n16) & 1u))        { a0 = -1.0e9f; keep0 = 0.0f; }
        if (!((w >> (n16 + 16)) & 1u)) { a1 = -1.0e9f; keep1 = 0.0f; }
      }
      float mx = fmaxf(a0, a1);
#pragma unroll
      for (int d = 8; d >= 1; d >>= 1) mx = fmaxf(mx, __shfl_xor(mx, d, 32));
      float mnew = fmaxf(mrow[r], mx);
      float c  = __expf(mrow[r] - mnew);
      float p0 = keep0 * __expf(a0 - mnew);
      float p1 = keep1 * __expf(a1 - mnew);
      float rs = p0 + p1;
#pragma unroll
      for (int d = 8; d >= 1; d >>= 1) rs += __shfl_xor(rs, d, 32);
      lrow[r] = lrow[r] * c + rs;
      mrow[r] = mnew;
      corr[r] = c;
      Pt[wave][r + 8 * g][n16]      = (half_t)p0;
      Pt[wave][r + 8 * g][n16 + 16] = (half_t)p1;
    }
#pragma unroll
    for (int t = 0; t < 4; ++t)
#pragma unroll
      for (int r = 0; r < 8; ++r) o[t][r] *= corr[r];

    __syncthreads();   // Vt + Pt visible

    v16h pa;
    {
      const half_t* p = &Pt[wave][n16][0];
      pa = frag_cat(*(const v8h*)(p + 16 * g), *(const v8h*)(p + 16 * g + 8));
    }
#pragma unroll
    for (int t = 0; t < 4; ++t) {
      const half_t* p = &Vt[t * 16 + n16][0];
      v16h bv = frag_cat(*(const v8h*)(p + 16 * g), *(const v8h*)(p + 16 * g + 8));
      o[t] = wmma_f16(pa, bv, o[t]);
    }
    __syncthreads();   // protect Vt before next iteration's staging
  }

#pragma unroll
  for (int r = 0; r < 8; ++r) {
    float inv = 1.0f / lrow[r];
    int row = qBase + r + 8 * g;
#pragma unroll
    for (int t = 0; t < 4; ++t)
      Out[(size_t)row * LD + hOff + t * 16 + n16] = (half_t)(o[t][r] * inv);
  }
}

// ---------------------------------------------------------------------------
// Fused residual + LayerNorm, optional f32 and f16 (strided) outputs.
// ---------------------------------------------------------------------------
__global__ __launch_bounds__(256) void k_ln_res(
    const float* __restrict__ a, const float* __restrict__ res,
    const float* __restrict__ gamma, const float* __restrict__ beta,
    float* __restrict__ out32, half_t* __restrict__ out16, int ld16,
    int off16) {
  __shared__ float sred[16];
  const int row = blockIdx.x;
  const int tid = threadIdx.x;
  const float* ap = a + (size_t)row * HID;
  const float* rp = res + (size_t)row * HID;
  float y0 = ap[tid] + rp[tid];
  float y1 = ap[tid + 256] + rp[tid + 256];
  float s = y0 + y1, sq = y0 * y0 + y1 * y1;
#pragma unroll
  for (int d = 16; d >= 1; d >>= 1) {
    s  += __shfl_xor(s,  d, 32);
    sq += __shfl_xor(sq, d, 32);
  }
  int wave = tid >> 5, lane = tid & 31;
  if (lane == 0) { sred[wave] = s; sred[8 + wave] = sq; }
  __syncthreads();
  if (tid == 0) {
    float ts = 0.0f, tq = 0.0f;
    for (int i = 0; i < 8; ++i) { ts += sred[i]; tq += sred[8 + i]; }
    sred[0] = ts; sred[8] = tq;
  }
  __syncthreads();
  float mean = sred[0] * (1.0f / HID);
  float var  = sred[8] * (1.0f / HID) - mean * mean;
  float rs   = rsqrtf(var + 1e-5f);
  float v0 = (y0 - mean) * rs * gamma[tid] + beta[tid];
  float v1 = (y1 - mean) * rs * gamma[tid + 256] + beta[tid + 256];
  if (out32) {
    out32[(size_t)row * HID + tid]       = v0;
    out32[(size_t)row * HID + tid + 256] = v1;
  }
  if (out16) {
    out16[(size_t)row * ld16 + off16 + tid]       = (half_t)v0;
    out16[(size_t)row * ld16 + off16 + tid + 256] = (half_t)v1;
  }
}

// ---------------------------------------------------------------------------
// Host-side orchestration
// ---------------------------------------------------------------------------
static void run_gemm(bool relu, bool out16, const half_t* A, const half_t* Bt,
                     const float* bias, float* C32, half_t* C16, int M, int N,
                     int K, int lda, int ldb, int ldc, hipStream_t s) {
  dim3 grid(N / 64, (M + 127) / 128);
  dim3 block(256);
  if (relu) {
    if (out16) k_gemm<true, true><<<grid, block, 0, s>>>(A, Bt, bias, C32, C16, M, N, K, lda, ldb, ldc);
    else       k_gemm<true, false><<<grid, block, 0, s>>>(A, Bt, bias, C32, C16, M, N, K, lda, ldb, ldc);
  } else {
    if (out16) k_gemm<false, true><<<grid, block, 0, s>>>(A, Bt, bias, C32, C16, M, N, K, lda, ldb, ldc);
    else       k_gemm<false, false><<<grid, block, 0, s>>>(A, Bt, bias, C32, C16, M, N, K, lda, ldb, ldc);
  }
}

extern "C" void kernel_launch(void* const* d_in, const int* in_sizes, int n_in,
                              void* d_out, int out_size, void* d_ws,
                              size_t ws_size, hipStream_t stream) {
  (void)in_sizes; (void)n_in; (void)out_size; (void)ws_size;

  const float* x      = (const float*)d_in[0];
  const float* lq_w   = (const float*)d_in[1];
  const float* lq_b   = (const float*)d_in[2];
  const float* lk_w   = (const float*)d_in[3];
  const float* lk_b   = (const float*)d_in[4];
  const float* lv_w   = (const float*)d_in[5];
  const float* lv_b   = (const float*)d_in[6];
  const float* lo_w   = (const float*)d_in[7];
  const float* lo_b   = (const float*)d_in[8];
  const float* gq_w   = (const float*)d_in[9];
  const float* gq_b   = (const float*)d_in[10];
  const float* gk_w   = (const float*)d_in[11];
  const float* gk_b   = (const float*)d_in[12];
  const float* gv_w   = (const float*)d_in[13];
  const float* gv_b   = (const float*)d_in[14];
  const float* go_w   = (const float*)d_in[15];
  const float* go_b   = (const float*)d_in[16];
  const float* centro = (const float*)d_in[17];
  const float* comb_w = (const float*)d_in[18];
  const float* comb_b = (const float*)d_in[19];
  const float* f1_w   = (const float*)d_in[20];
  const float* f1_b   = (const float*)d_in[21];
  const float* f2_w   = (const float*)d_in[22];
  const float* f2_b   = (const float*)d_in[23];
  const float* l_ln_g = (const float*)d_in[24];
  const float* l_ln_b = (const float*)d_in[25];
  const float* g_ln_g = (const float*)d_in[26];
  const float* g_ln_b = (const float*)d_in[27];
  const float* h_ln_g = (const float*)d_in[28];
  const float* h_ln_b = (const float*)d_in[29];
  const float* f_ln_g = (const float*)d_in[30];
  const float* f_ln_b = (const float*)d_in[31];
  const int*   eidx   = (const int*)d_in[32];
  float*       out    = (float*)d_out;

  char* wsp = (char*)d_ws;
  auto alloc = [&](size_t bytes) -> void* {
    void* p = (void*)wsp;
    wsp += (bytes + 255) & ~(size_t)255;
    return p;
  };
  uint32_t* adjBits  = (uint32_t*)alloc((size_t)NNODES * MASKW * 4);
  uint32_t* maskBits = (uint32_t*)alloc((size_t)NNODES * MASKW * 4);
  half_t* xh    = (half_t*)alloc((size_t)NNODES * HID * 2);
  half_t* ch    = (half_t*)alloc((size_t)NC * HID * 2);
  half_t* lqT   = (half_t*)alloc((size_t)HID * HID * 2);
  half_t* lkT   = (half_t*)alloc((size_t)HID * HID * 2);
  half_t* lvT   = (half_t*)alloc((size_t)HID * HID * 2);
  half_t* loT   = (half_t*)alloc((size_t)HID * HID * 2);
  half_t* gqT   = (half_t*)alloc((size_t)HID * HID * 2);
  half_t* gkT   = (half_t*)alloc((size_t)HID * HID * 2);
  half_t* gvT   = (half_t*)alloc((size_t)HID * HID * 2);
  half_t* goT   = (half_t*)alloc((size_t)HID * HID * 2);
  half_t* combT = (half_t*)alloc((size_t)HID * 2 * HID * 2);
  half_t* f1T   = (half_t*)alloc((size_t)4 * HID * HID * 2);
  half_t* f2T   = (half_t*)alloc((size_t)HID * 4 * HID * 2);
  half_t* Qh    = (half_t*)alloc((size_t)NNODES * HID * 2);
  half_t* Kh    = (half_t*)alloc((size_t)NNODES * HID * 2);
  half_t* Vh    = (half_t*)alloc((size_t)NNODES * HID * 2);
  half_t* Qgh   = (half_t*)alloc((size_t)NNODES * HID * 2);
  half_t* Kch   = (half_t*)alloc((size_t)NC * HID * 2);
  half_t* Vch   = (half_t*)alloc((size_t)NC * HID * 2);
  half_t* attO  = (half_t*)alloc((size_t)NNODES * HID * 2);
  half_t* gattO = (half_t*)alloc((size_t)NNODES * HID * 2);
  half_t* concatH = (half_t*)alloc((size_t)NNODES * 2 * HID * 2);
  float*  tmp32 = (float*)alloc((size_t)NNODES * HID * 4);
  float*  hyb32 = (float*)alloc((size_t)NNODES * HID * 4);
  half_t* hybH  = (half_t*)alloc((size_t)NNODES * HID * 2);
  half_t* f1H   = (half_t*)alloc((size_t)NNODES * 4 * HID * 2);

  // ---- 1) adjacency bitsets + 2-hop reach mask
  k_zero_u32<<<1024, 256, 0, stream>>>(adjBits, NNODES * MASKW);
  k_edges<<<NEDGES / 256, 256, 0, stream>>>(eidx, adjBits);
  k_diag<<<NNODES / 256, 256, 0, stream>>>(adjBits);
  k_reach<<<NNODES, 128, 0, stream>>>(adjBits, maskBits);

  // ---- 2) precision conversion + weight transposes
  k_cvt16<<<2048, 256, 0, stream>>>(x, xh, (long)NNODES * HID);
  k_cvt16<<<128, 256, 0, stream>>>(centro, ch, (long)NC * HID);
  k_transpose16<<<1024, 256, 0, stream>>>(lq_w, lqT, HID, HID);
  k_transpose16<<<1024, 256, 0, stream>>>(lk_w, lkT, HID, HID);
  k_transpose16<<<1024, 256, 0, stream>>>(lv_w, lvT, HID, HID);
  k_transpose16<<<1024, 256, 0, stream>>>(lo_w, loT, HID, HID);
  k_transpose16<<<1024, 256, 0, stream>>>(gq_w, gqT, HID, HID);
  k_transpose16<<<1024, 256, 0, stream>>>(gk_w, gkT, HID, HID);
  k_transpose16<<<1024, 256, 0, stream>>>(gv_w, gvT, HID, HID);
  k_transpose16<<<1024, 256, 0, stream>>>(go_w, goT, HID, HID);
  k_transpose16<<<2048, 256, 0, stream>>>(comb_w, combT, 2 * HID, HID);
  k_transpose16<<<2048, 256, 0, stream>>>(f1_w, f1T, HID, 4 * HID);
  k_transpose16<<<2048, 256, 0, stream>>>(f2_w, f2T, 4 * HID, HID);

  // ---- 3) QKV projections (WMMA GEMMs, f16 out)
  run_gemm(false, true, xh, lqT, lq_b, nullptr, Qh,  NNODES, HID, HID, HID, HID, HID, stream);
  run_gemm(false, true, xh, lkT, lk_b, nullptr, Kh,  NNODES, HID, HID, HID, HID, HID, stream);
  run_gemm(false, true, xh, lvT, lv_b, nullptr, Vh,  NNODES, HID, HID, HID, HID, HID, stream);
  run_gemm(false, true, xh, gqT, gq_b, nullptr, Qgh, NNODES, HID, HID, HID, HID, HID, stream);
  run_gemm(false, true, ch, gkT, gk_b, nullptr, Kch, NC, HID, HID, HID, HID, HID, stream);
  run_gemm(false, true, ch, gvT, gv_b, nullptr, Vch, NC, HID, HID, HID, HID, HID, stream);

  // ---- 4) local masked flash attention + global centroid attention
  {
    dim3 grid(NNODES / 64, HEADS);
    k_flash<<<grid, 128, 0, stream>>>(Qh, Kh, Vh, maskBits, attO, NNODES, MASKW, ATT_SCALE);
    k_flash<<<grid, 128, 0, stream>>>(Qgh, Kch, Vch, nullptr, gattO, NC, 0, ATT_SCALE);
  }

  // ---- 5) output projections + residual LN -> concat buffer (f16)
  run_gemm(false, false, attO, loT, lo_b, tmp32, nullptr, NNODES, HID, HID, HID, HID, HID, stream);
  k_ln_res<<<NNODES, 256, 0, stream>>>(tmp32, x, l_ln_g, l_ln_b, nullptr, concatH, 2 * HID, 0);
  run_gemm(false, false, gattO, goT, go_b, tmp32, nullptr, NNODES, HID, HID, HID, HID, HID, stream);
  k_ln_res<<<NNODES, 256, 0, stream>>>(tmp32, x, g_ln_g, g_ln_b, nullptr, concatH, 2 * HID, HID);

  // ---- 6) combine (K=1024 GEMM + ReLU) + LN
  run_gemm(true, false, concatH, combT, comb_b, tmp32, nullptr, NNODES, HID, 2 * HID, 2 * HID, 2 * HID, HID, stream);
  k_ln_res<<<NNODES, 256, 0, stream>>>(tmp32, x, h_ln_g, h_ln_b, hyb32, hybH, HID, 0);

  // ---- 7) FFN + final LN
  run_gemm(true, true, hybH, f1T, f1_b, nullptr, f1H, NNODES, 4 * HID, HID, HID, HID, 4 * HID, stream);
  run_gemm(false, false, f1H, f2T, f2_b, tmp32, nullptr, NNODES, HID, 4 * HID, 4 * HID, 4 * HID, HID, stream);
  k_ln_res<<<NNODES, 256, 0, stream>>>(tmp32, hyb32, f_ln_g, f_ln_b, out, nullptr, 0, 0);
}
